// MultiHeadAttn_58514634441417
// MI455X (gfx1250) — matrix operand order repaired and checked by reading, not compiled, not run on hardware
//
#include <hip/hip_runtime.h>
#include <hip/hip_bf16.h>

// ---------------------------------------------------------------------------
// MultiHeadAttn (conv-qkv attention) for gfx1250 (MI455X, CDNA5, wave32).
// All heavy math on v_wmma_f32_16x16x32_bf16 (f32 accumulate).
// All WMMA operands are fragment-major so each lane loads 32B with b128 ops.
// Conv: branch-free cndmask padding (precomputed tap mask), LDS double buffer.
// ---------------------------------------------------------------------------

typedef __attribute__((ext_vector_type(16))) __bf16 v16bf;
typedef __attribute__((ext_vector_type(8)))  float  v8f;

#define CDIM 1024                 // channels == tokens
#define NHEAD 16
#define HDIM 64                   // head dim
#define BATCH 8
#define KCONV (9 * CDIM)          // im2col K = 9216
#define KCHUNKS (KCONV / 32)      // 288
#define MPIX (BATCH * 32 * 32)    // 8192 GEMM rows

__device__ __forceinline__ __bf16 bf_bits(unsigned short u) {
  return __builtin_bit_cast(__bf16, u);
}

// ---------------- fragment coordinate maps (bf16, contiguous k = 16h + e) ----------------
// A 16x32: lane = m + 16*(k>>4); elem e = k & 15.
__device__ __forceinline__ void a_coords(int m, int k, int& lane, int& e) {
  lane = m + ((k >> 4) << 4);
  e = k & 15;
}
// B 32x16: lane = n + 16*(k>>4); elem e = k & 15.
__device__ __forceinline__ void b_coords(int k, int n, int& lane, int& e) {
  lane = n + ((k >> 4) << 4);
  e = k & 15;
}
// C/D 16x16 f32: element j -> row M = j + 8*(lane>>4), col N = lane&15.

// ---------------------------- reformat kernels ------------------------------
__global__ __launch_bounds__(256) void cast_bf16_kernel(
    const float* __restrict__ in, __bf16* __restrict__ out, int n) {
  int i = blockIdx.x * 256 + threadIdx.x;
  if (i < n) out[i] = (__bf16)in[i];
}

// W [O=1024][I=1024][3][3] fp32 -> fragment-major WtP[kc(288)][ntile(64)][lane][16]
// GEMM K order: k = tap*1024 + i  (tap-major: each 32-chunk = one tap, 32 chans)
__global__ __launch_bounds__(256) void wconv_permute_kernel(
    const float* __restrict__ Win, __bf16* __restrict__ WtP) {
  int idx = blockIdx.x * 256 + threadIdx.x;
  if (idx >= KCONV * CDIM) return;
  const int e     = idx & 15;
  const int lane  = (idx >> 4) & 31;
  const int ntile = (idx >> 9) & 63;
  const int kc    = idx >> 15;
  const int kl = ((lane >> 4) << 4) + e;      // B map inverse
  const int k  = kc * 32 + kl;
  const int i  = k & 1023, tap = k >> 10;
  const int n  = ntile * 16 + (lane & 15);
  WtP[idx] = (__bf16)Win[(long)n * KCONV + i * 9 + tap];
}

// Wo [o][p] fp32 -> fragment-major WoP[kc(32)][ntile(64)][lane][16] (B = Wo^T)
__global__ __launch_bounds__(256) void wo_permute_kernel(
    const float* __restrict__ Wo, __bf16* __restrict__ WoP) {
  int idx = blockIdx.x * 256 + threadIdx.x;
  if (idx >= CDIM * CDIM) return;
  const int e     = idx & 15;
  const int lane  = (idx >> 4) & 31;
  const int ntile = (idx >> 9) & 63;
  const int kc    = idx >> 15;
  const int p = kc * 32 + ((lane >> 4) << 4) + e;
  const int o = ntile * 16 + (lane & 15);
  WoP[idx] = (__bf16)Wo[o * CDIM + p];
}

// ---------------- fragment-major scatter addresses for Q/K/V ----------------
// mode 0: Q as A-fragments [bh][qstrip(64)][ddchunk(2)][lane][16]
// mode 1: K as B-fragments [bh][stile(64)][ddchunk(2)][lane][16]  (n=s, k=dd)
// mode 2: V as B-fragments [bh][schunk(32)][dtile(4)][lane][16]   (n=dd, k=s)
__device__ __forceinline__ long frag_addr(int mode, int bh, int dd, int o) {
  int lane, e;
  if (mode == 0) {
    a_coords(o & 15, dd & 31, lane, e);
    return ((((long)bh * 64 + (o >> 4)) * 2 + (dd >> 5)) * 32 + lane) * 16 + e;
  } else if (mode == 1) {
    b_coords(dd & 31, o & 15, lane, e);
    return ((((long)bh * 64 + (o >> 4)) * 2 + (dd >> 5)) * 32 + lane) * 16 + e;
  } else {
    b_coords(o & 31, dd & 15, lane, e);
    return ((((long)bh * 32 + (o >> 5)) * 4 + (dd >> 4)) * 32 + lane) * 16 + e;
  }
}

// ----------------------- conv3x3 as implicit-im2col WMMA --------------------
// M=8192 pixels, N=1024 out-ch, K=9216. 64x64 tile per WG, 4 waves (16x64
// strip each). A double-buffered in LDS in fragment-major order (b128 reads,
// branch-free cndmask padding via precomputed 9-bit tap validity mask);
// B fragments register-prefetched directly from fragment-major weights.
__global__ __launch_bounds__(128) void conv3x3_wmma_kernel(
    const unsigned short* __restrict__ Xu,  // bf16 bits [B,1024,32,32]
    const __bf16* __restrict__ WtP,         // fragment-major weights
    const float*  __restrict__ bias,        // [1024]
    __bf16* __restrict__ Out, int mode) {
  __shared__ __align__(32) __bf16 Af[2][4 * 32 * 16];  // double-buffered tile
  const int tid  = threadIdx.x;
  const int lane = tid & 31, wave = tid >> 5;
  const int n0 = blockIdx.x * 64;
  const int m0 = blockIdx.y * 64;
  const int nt0 = n0 >> 4;

  // loop-invariant staging state: 16 elements per thread
  int gOff[16];  // (b*1024 + cl)*1024 + y*32 + x
  int pk[16];    // [15:0] fragment-major LDS index, [24:16] 9-bit tap validity
#pragma unroll
  for (int it = 0; it < 16; ++it) {
    const int ee = it * 128 + tid;
    const int cl = ee >> 6, m = ee & 63;
    const int g = m0 + m;
    const int b = g >> 10, pp = g & 1023;
    const int y = pp >> 5, x = pp & 31;
    gOff[it] = (b * CDIM + cl) * CDIM + pp;
    int la, el;
    a_coords(m & 15, cl, la, el);
    int m9 = 0;
#pragma unroll
    for (int tap = 0; tap < 9; ++tap) {
      const int dy = (tap >= 3) + (tap >= 6) - 1;
      const int dx = tap - (dy + 1) * 3 - 1;
      const bool inb =
          ((unsigned)(y + dy) < 32u) && ((unsigned)(x + dx) < 32u);
      m9 |= (inb ? 1 : 0) << tap;
    }
    pk[it] = (((m >> 4) * 32 + la) * 16 + el) | (m9 << 16);
  }

  // branch-free padded stage of K-chunk kcs into its parity buffer
  auto stage = [&](int kcs) {
    const int tap = kcs >> 5;
    const int dy = (tap >= 3) + (tap >= 6) - 1;              // -1,0,1 (no div)
    const int dx = tap - (dy + 1) * 3 - 1;
    const int sOff = (kcs & 31) * 32 * CDIM + dy * 32 + dx;  // uniform scalar
    const int mbit = 1 << (16 + tap);                        // uniform scalar
    __bf16* dst = &Af[kcs & 1][0];
#pragma unroll
    for (int it = 0; it < 16; ++it) {
      const bool inb = (pk[it] & mbit) != 0;                 // and + cmp
      const int a2 = inb ? (gOff[it] + sOff) : gOff[it];     // cndmask address
      const unsigned short r = Xu[a2];                       // uncond. load
      dst[pk[it] & 0xFFFF] = bf_bits(inb ? r : (unsigned short)0);
    }
  };

  v8f acc[4];
#pragma unroll
  for (int t = 0; t < 4; ++t)
#pragma unroll
    for (int j = 0; j < 8; ++j) acc[t][j] = 0.f;

  // prime B fragments (kc=0) and A buffer 0
  v16bf Bcur[4];
#pragma unroll
  for (int nt = 0; nt < 4; ++nt)
    Bcur[nt] = *reinterpret_cast<const v16bf*>(
        WtP + (((long)(nt0 + nt)) * 32 + lane) * 16);
  stage(0);
  __syncthreads();

  for (int kc = 0; kc < KCHUNKS; ++kc) {
    if (kc + 1 < KCHUNKS) stage(kc + 1);     // overlaps with WMMAs below
    const v16bf a = *reinterpret_cast<const v16bf*>(
        &Af[kc & 1][(wave * 32 + lane) * 16]);
#pragma unroll
    for (int nt = 0; nt < 4; ++nt) {
      const v16bf bu = Bcur[nt];
      if (kc + 1 < KCHUNKS)                  // register prefetch next B frag
        Bcur[nt] = *reinterpret_cast<const v16bf*>(
            WtP + (((long)(kc + 1) * 64 + nt0 + nt) * 32 + lane) * 16);
      acc[nt] = __builtin_amdgcn_wmma_f32_16x16x32_bf16(
          false, a, false, bu, (short)0, acc[nt], false, false);
    }
    __syncthreads();                         // one barrier per K-step
  }

  // bias + scatter into fragment-major attention layouts
  const int half = lane >> 4, nn = lane & 15;
#pragma unroll
  for (int nt = 0; nt < 4; ++nt) {
    const int o = n0 + nt * 16 + nn;         // output channel == token
    const float bo = bias[o];
#pragma unroll
    for (int j = 0; j < 8; ++j) {
      const int m  = wave * 16 + j + half * 8;
      const int g  = m0 + m;
      const int b  = g >> 10;
      const int pp = g & 1023;               // spatial feature index
      const int head = pp >> 6, dd = pp & 63;
      Out[frag_addr(mode, b * NHEAD + head, dd, o)] = (__bf16)(acc[nt][j] + bo);
    }
  }
}

// -------------------- flash attention, 64-query tile per WG -----------------
// Q fragments live in VGPRs for the whole pass; K/V fragments are direct
// 32B global loads (L2-resident); only P round-trips through per-wave LDS.
__global__ __launch_bounds__(128) void attn_wmma_kernel(
    const __bf16* __restrict__ QP,   // A-frag major
    const __bf16* __restrict__ KP,   // B-frag major (n=s, k=dd)
    const __bf16* __restrict__ VP,   // B-frag major (n=dd, k=s)
    const int*    __restrict__ mask, // [B][1024][1024]
    __bf16* __restrict__ OP) {       // proj-A-frag major [strip(512)][kc(32)][lane][16]
  __shared__ __align__(32) __bf16 Pf[4][2][32][16];  // per-wave P staging
  const int tid = threadIdx.x, lane = tid & 31, wave = tid >> 5;
  const int q0 = blockIdx.x * 64;
  const int bh = blockIdx.y;
  const int b = bh >> 4, h = bh & 15;
  const int half = lane >> 4, nn = lane & 15;

  // Q fragments for this wave's 16-row strip (resident across the s-loop)
  const int qstrip = (q0 >> 4) + wave;
  v16bf aQ[2];
#pragma unroll
  for (int c = 0; c < 2; ++c)
    aQ[c] = *reinterpret_cast<const v16bf*>(
        QP + ((((long)bh * 64 + qstrip) * 2 + c) * 32 + lane) * 16);

  float mrow[8], lrow[8];
#pragma unroll
  for (int j = 0; j < 8; ++j) { mrow[j] = -3.0e38f; lrow[j] = 0.f; }
  v8f Oacc[4];
#pragma unroll
  for (int t = 0; t < 4; ++t)
#pragma unroll
    for (int j = 0; j < 8; ++j) Oacc[t][j] = 0.f;

  for (int s0 = 0; s0 < CDIM; s0 += 64) {
    // ---- S = Q K^T : 2 dd-chunks x 4 s-tiles ----
    v8f S[4];
#pragma unroll
    for (int t = 0; t < 4; ++t)
#pragma unroll
      for (int j = 0; j < 8; ++j) S[t][j] = 0.f;
#pragma unroll
    for (int c = 0; c < 2; ++c) {
#pragma unroll
      for (int nt = 0; nt < 4; ++nt) {
        const int stile = (s0 >> 4) + nt;
        const v16bf bK = *reinterpret_cast<const v16bf*>(
            KP + ((((long)bh * 64 + stile) * 2 + c) * 32 + lane) * 16);
        S[nt] = __builtin_amdgcn_wmma_f32_16x16x32_bf16(
            false, aQ[c], false, bK, (short)0, S[nt], false, false);
      }
    }
    if (s0 + 64 < CDIM)
      __builtin_prefetch(
          KP + ((((long)bh * 64 + ((s0 + 64) >> 4)) * 2) * 32 + lane) * 16, 0, 0);

    // ---- scale + mask ----
#pragma unroll
    for (int nt = 0; nt < 4; ++nt) {
      const int s = s0 + nt * 16 + nn;
#pragma unroll
      for (int j = 0; j < 8; ++j) {
        const int t = q0 + wave * 16 + j + half * 8;
        float v = S[nt][j] * 0.125f;  // 1/sqrt(64)
        if (mask[((long)b << 20) + ((long)t << 10) + s] == 0) v = -1.0e9f;
        S[nt][j] = v;
      }
    }
    // ---- online softmax (rows span 16 lanes: xor-reduce 1..8) ----
#pragma unroll
    for (int j = 0; j < 8; ++j) {
      float mx = fmaxf(fmaxf(S[0][j], S[1][j]), fmaxf(S[2][j], S[3][j]));
#pragma unroll
      for (int off = 1; off < 16; off <<= 1)
        mx = fmaxf(mx, __shfl_xor(mx, off, 32));
      const float mnew = fmaxf(mrow[j], mx);
      const float alpha = __expf(mrow[j] - mnew);
      mrow[j] = mnew;
      float r = 0.f;
#pragma unroll
      for (int nt = 0; nt < 4; ++nt) {
        const float p = __expf(S[nt][j] - mnew);
        S[nt][j] = p;
        r += p;
      }
#pragma unroll
      for (int off = 1; off < 16; off <<= 1) r += __shfl_xor(r, off, 32);
      lrow[j] = lrow[j] * alpha + r;
#pragma unroll
      for (int nt = 0; nt < 4; ++nt) Oacc[nt][j] *= alpha;
    }
    // ---- stage P into per-wave fragment-major LDS (intra-wave in-order) ----
#pragma unroll
    for (int nt = 0; nt < 4; ++nt) {
#pragma unroll
      for (int j = 0; j < 8; ++j) {
        const int r = j + half * 8;          // row within strip
        const int c = nt * 16 + nn;          // s-local 0..63
        int la, el;
        a_coords(r, c & 31, la, el);
        Pf[wave][c >> 5][la][el] = (__bf16)S[nt][j];
      }
    }
    // ---- O += P V : 2 s-chunks x 4 d-tiles ----
#pragma unroll
    for (int c = 0; c < 2; ++c) {
      const v16bf aP = *reinterpret_cast<const v16bf*>(&Pf[wave][c][lane][0]);
      const int schunk = (s0 >> 5) + c;
#pragma unroll
      for (int dt = 0; dt < 4; ++dt) {
        const v16bf bV = *reinterpret_cast<const v16bf*>(
            VP + ((((long)bh * 32 + schunk) * 4 + dt) * 32 + lane) * 16);
        Oacc[dt] = __builtin_amdgcn_wmma_f32_16x16x32_bf16(
            false, aP, false, bV, (short)0, Oacc[dt], false, false);
      }
    }
  }

  // ---- normalize; emit directly as projection-A fragments ----
#pragma unroll
  for (int dt = 0; dt < 4; ++dt) {
#pragma unroll
    for (int j = 0; j < 8; ++j) {
      const int t = q0 + wave * 16 + j + half * 8;
      const long mrowg = (long)b * CDIM + t;          // GEMM row 0..8191
      const int p = h * HDIM + dt * 16 + nn;          // feature 0..1023
      int la, el;
      a_coords((int)(mrowg & 15), p & 31, la, el);
      const long addr = ((((mrowg >> 4) * 32 + (p >> 5)) * 32 + la) * 16 + el);
      OP[addr] = (__bf16)(Oacc[dt][j] / lrow[j]);
    }
  }
}

// ------------------------- output projection GEMM ---------------------------
// C[8192,1024] = A(frag-major bf16) * WoP(frag-major bf16) + bias. Zero LDS.
__global__ __launch_bounds__(128) void proj_wmma_kernel(
    const __bf16* __restrict__ AP, const __bf16* __restrict__ WoP,
    const float* __restrict__ bias, float* __restrict__ C) {
  const int tid = threadIdx.x, lane = tid & 31, wave = tid >> 5;
  const int n0 = blockIdx.x * 64, m0 = blockIdx.y * 64;
  const int strip = (m0 >> 4) + wave, nt0 = n0 >> 4;
  v8f acc[4];
#pragma unroll
  for (int t = 0; t < 4; ++t)
#pragma unroll
    for (int j = 0; j < 8; ++j) acc[t][j] = 0.f;

  for (int kc = 0; kc < 32; ++kc) {
    const v16bf a = *reinterpret_cast<const v16bf*>(
        AP + (((long)strip * 32 + kc) * 32 + lane) * 16);
#pragma unroll
    for (int nt = 0; nt < 4; ++nt) {
      const v16bf bfr = *reinterpret_cast<const v16bf*>(
          WoP + (((long)kc * 64 + nt0 + nt) * 32 + lane) * 16);
      acc[nt] = __builtin_amdgcn_wmma_f32_16x16x32_bf16(
          false, a, false, bfr, (short)0, acc[nt], false, false);
    }
  }
  const int half = lane >> 4, nn = lane & 15;
#pragma unroll
  for (int nt = 0; nt < 4; ++nt) {
    const int o = n0 + nt * 16 + nn;
    const float bo = bias[o];
#pragma unroll
    for (int j = 0; j < 8; ++j) {
      const int m = wave * 16 + j + half * 8;
      C[(long)(m0 + m) * CDIM + o] = acc[nt][j] + bo;
    }
  }
}

// --------------------------------- driver -----------------------------------
extern "C" void kernel_launch(void* const* d_in, const int* in_sizes, int n_in,
                              void* d_out, int out_size, void* d_ws, size_t ws_size,
                              hipStream_t stream) {
  const float* q    = (const float*)d_in[0];
  const float* k    = (const float*)d_in[1];
  const float* v    = (const float*)d_in[2];
  const float* Wq   = (const float*)d_in[3];
  const float* bq   = (const float*)d_in[4];
  const float* Wk   = (const float*)d_in[5];
  const float* bk   = (const float*)d_in[6];
  const float* Wv   = (const float*)d_in[7];
  const float* bv   = (const float*)d_in[8];
  const float* Wo   = (const float*)d_in[9];
  const float* bo   = (const float*)d_in[10];
  const int*   mask = (const int*)d_in[11];
  float* out = (float*)d_out;

  char* ws = (char*)d_ws;
  size_t off = 0;
  auto alloc = [&](size_t bytes) -> void* {
    off = (off + 255) & ~(size_t)255;
    void* p = ws + off;
    off += bytes;
    return p;
  };
  const size_t nQKV = (size_t)BATCH * CDIM * 32 * 32;  // 8388608
  __bf16* Xq  = (__bf16*)alloc(nQKV * 2);
  __bf16* Xk  = (__bf16*)alloc(nQKV * 2);
  __bf16* Xv  = (__bf16*)alloc(nQKV * 2);
  __bf16* WqP = (__bf16*)alloc((size_t)KCONV * CDIM * 2);
  __bf16* WkP = (__bf16*)alloc((size_t)KCONV * CDIM * 2);
  __bf16* WvP = (__bf16*)alloc((size_t)KCONV * CDIM * 2);
  __bf16* QP  = (__bf16*)alloc(nQKV * 2);
  __bf16* KP  = (__bf16*)alloc(nQKV * 2);
  __bf16* VP  = (__bf16*)alloc(nQKV * 2);
  __bf16* OP  = (__bf16*)alloc(nQKV * 2);
  __bf16* WoP = (__bf16*)alloc((size_t)CDIM * CDIM * 2);
  (void)ws_size; (void)in_sizes; (void)n_in; (void)out_size;

  // 1) precision conversion + fragment-major weight permutes
  {
    const int n = (int)nQKV;
    const int g = (n + 255) / 256;
    cast_bf16_kernel<<<g, 256, 0, stream>>>(q, Xq, n);
    cast_bf16_kernel<<<g, 256, 0, stream>>>(k, Xk, n);
    cast_bf16_kernel<<<g, 256, 0, stream>>>(v, Xv, n);
    const int gw = (KCONV * CDIM + 255) / 256;
    wconv_permute_kernel<<<gw, 256, 0, stream>>>(Wq, WqP);
    wconv_permute_kernel<<<gw, 256, 0, stream>>>(Wk, WkP);
    wconv_permute_kernel<<<gw, 256, 0, stream>>>(Wv, WvP);
    wo_permute_kernel<<<(CDIM * CDIM + 255) / 256, 256, 0, stream>>>(Wo, WoP);
  }
  // 2) conv projections -> fragment-major Q/K/V
  {
    dim3 grid(CDIM / 64, MPIX / 64);  // 16 x 128
    conv3x3_wmma_kernel<<<grid, 128, 0, stream>>>((const unsigned short*)Xq, WqP, bq, QP, 0);
    conv3x3_wmma_kernel<<<grid, 128, 0, stream>>>((const unsigned short*)Xk, WkP, bk, KP, 1);
    conv3x3_wmma_kernel<<<grid, 128, 0, stream>>>((const unsigned short*)Xv, WvP, bv, VP, 2);
  }
  // 3) masked flash attention -> projection-A fragments
  {
    dim3 grid(CDIM / 64, BATCH * NHEAD);  // 16 x 128
    attn_wmma_kernel<<<grid, 128, 0, stream>>>(QP, KP, VP, mask, OP);
  }
  // 4) output projection + bias (fp32 out)
  {
    dim3 grid(CDIM / 64, MPIX / 64);  // 16 x 128
    proj_wmma_kernel<<<grid, 128, 0, stream>>>(OP, WoP, bo, out);
  }
}
